// Encoder_EmbTreeRNN_80874234184081
// MI455X (gfx1250) — compile-verified, matrix operand order checked by reading, not run on hardware
//
#include <hip/hip_runtime.h>

// ---------------------------------------------------------------------------
// Tree-LSTM encoder for MI455X (gfx1250, wave32, WMMA bf16 -> f32 accumulate)
// ---------------------------------------------------------------------------

constexpr int Bc   = 64;      // batch (trees)
constexpr int Dd   = 11;      // tree depth
constexpr int Tt   = 2047;    // nodes per tree
constexpr int Hh   = 256;
constexpr int Ee   = 256;
constexpr int Ll   = 5;
constexpr int Vv   = 50000;
constexpr int LEAF = 1023;    // 2^(D-1)-1
constexpr int Nn   = Bc * Tt; // 131008

typedef __attribute__((ext_vector_type(16))) __bf16 v16bf;
typedef __attribute__((ext_vector_type(8)))  __bf16 v8bf;
typedef __attribute__((ext_vector_type(4)))  __bf16 v4bf;
typedef __attribute__((ext_vector_type(8)))  float  v8f;

union ABu { v16bf v; uint4 q[2]; };
union U8b { v8bf  b; uint4 q;    };
union U4b { v4bf  b; uint2 q;    };

__device__ __forceinline__ float sigf(float x) { return 1.0f / (1.0f + __expf(-x)); }

// --------------------------- fp32 -> bf16 convert ---------------------------
__global__ void cvt_kernel(const float* __restrict__ src, __bf16* __restrict__ dst, int n) {
  int i = blockIdx.x * blockDim.x + threadIdx.x;
  int stride = gridDim.x * blockDim.x;
  for (; i < n; i += stride) dst[i] = (__bf16)src[i];
}

// ------------------------------- leaf kernel --------------------------------
// One block = 16 leaf rows. Gather-sum 5 embeddings -> LDS bf16 A (16x256),
// GEMM vs W_iou^T (768 cols) with v_wmma_f32_16x16x32_bf16, then gates.
constexpr int LF_AS = 264;   // bf16 A row stride (4-bank skew, 16B aligned)
constexpr int LF_OS = 772;   // f32 out row stride (4-bank skew)

__global__ __launch_bounds__(256) void leaf_kernel(
    const int*    __restrict__ wordid,
    const __bf16* __restrict__ emb_bf,
    const __bf16* __restrict__ Wb,       // [768][256] bf16 row-major
    const float*  __restrict__ biou,     // [768]
    const float*  __restrict__ c0,       // [N][H]
    float* __restrict__ h_all,           // [N][H] (== tree_output)
    float* __restrict__ c_all)           // [N][H]
{
  extern __shared__ char smem[];
  __bf16* As = (__bf16*)smem;                                  // [16][LF_AS]
  float*  Os = (float*)(smem + 16 * LF_AS * sizeof(__bf16));   // [16][LF_OS]
  const int tid = threadIdx.x;
  const int li0 = blockIdx.x * 16;

  // phase 1: embedding gather + sum over L, store bf16 A-tile
  for (int task = tid; task < 16 * 32; task += 256) {
    int r = task >> 5, g = task & 31, col = g * 8;
    int li = li0 + r;
    int b = li >> 10, tl = li & 1023;
    size_t grow = (size_t)b * Tt + LEAF + tl;
    float a[8] = {0.f, 0.f, 0.f, 0.f, 0.f, 0.f, 0.f, 0.f};
#pragma unroll
    for (int l = 0; l < Ll; ++l) {
      int w = wordid[grow * Ll + l];
      U8b e; e.q = *(const uint4*)(emb_bf + (size_t)w * Ee + col);
#pragma unroll
      for (int i = 0; i < 8; ++i) a[i] += (float)e.b[i];
    }
    U8b o;
#pragma unroll
    for (int i = 0; i < 8; ++i) o.b[i] = (__bf16)a[i];
    *(uint4*)(As + r * LF_AS + col) = o.q;
  }
  __syncthreads();

  // phase 2: WMMA GEMM 16x768 (K=256). 8 waves x 6 N-tiles x 8 K-steps.
  {
    const int lane = tid & 31, wave = tid >> 5;
    const int nl  = lane & 15;
    const int hi8 = (lane >> 4) * 8;
    v8f acc[6];
#pragma unroll
    for (int j = 0; j < 6; ++j) acc[j] = v8f{0.f, 0.f, 0.f, 0.f, 0.f, 0.f, 0.f, 0.f};
    const __bf16* bp[6];
#pragma unroll
    for (int j = 0; j < 6; ++j) {
      int nt = wave * 6 + j;
      bp[j] = Wb + (size_t)(nt * 16 + nl) * 256;
    }
    for (int ks = 0; ks < 8; ++ks) {
      int kb = ks * 32;
      ABu a;
      a.q[0] = *(const uint4*)(As + nl * LF_AS + kb + hi8);
      a.q[1] = *(const uint4*)(As + nl * LF_AS + kb + 16 + hi8);
#pragma unroll
      for (int j = 0; j < 6; ++j) {
        ABu bb;
        bb.q[0] = *(const uint4*)(bp[j] + kb + hi8);
        bb.q[1] = *(const uint4*)(bp[j] + kb + 16 + hi8);
        acc[j] = __builtin_amdgcn_wmma_f32_16x16x32_bf16(
            false, a.v, false, bb.v, (short)0, acc[j], false, false);
      }
    }
#pragma unroll
    for (int j = 0; j < 6; ++j) {
      int nt = wave * 6 + j;
#pragma unroll
      for (int r = 0; r < 8; ++r)
        Os[(r + hi8) * LF_OS + nt * 16 + nl] = acc[j][r];
    }
  }
  __syncthreads();

  // phase 3: gates. cn = sig(i)*tanh(u) + c0 ; h = sig(o)*tanh(cn)
  for (int task = tid; task < 16 * 256; task += 256) {
    int r = task >> 8, j = task & 255;
    int li = li0 + r;
    int b = li >> 10, tl = li & 1023;
    size_t grow = (size_t)b * Tt + LEAF + tl;
    float i_ = Os[r * LF_OS + j]       + biou[j];
    float o_ = Os[r * LF_OS + 256 + j] + biou[256 + j];
    float u_ = Os[r * LF_OS + 512 + j] + biou[512 + j];
    float cn = sigf(i_) * tanhf(u_) + c0[grow * Hh + j];
    float hh = sigf(o_) * tanhf(cn);
    c_all[grow * Hh + j] = cn;
    h_all[grow * Hh + j] = hh;
  }
}

// ------------------------------- level kernel -------------------------------
// One block = 16 nodes of level d. A = concat(h_child0, h_child1) (16x512).
// 80 N-tiles: [0,32) -> U_f (512 cols), [32,80) -> U_iou (768 cols).
constexpr int LV_AS = 520;    // bf16 A row stride
constexpr int LV_OS = 1284;   // f32 out row stride

__global__ __launch_bounds__(256) void level_kernel(
    int s, int e, int nlevel,
    float* __restrict__ h_all, float* __restrict__ c_all,
    const __bf16* __restrict__ Uf,    // [512][512] bf16 row-major
    const __bf16* __restrict__ Uiou,  // [768][512] bf16 row-major
    const float*  __restrict__ Ufb,   // [512]
    const float*  __restrict__ biou)  // [768]
{
  extern __shared__ char smem[];
  __bf16* As = (__bf16*)smem;                                  // [16][LV_AS]
  float*  Os = (float*)(smem + 16 * LV_AS * sizeof(__bf16));   // [16][LV_OS]
  const int tid  = threadIdx.x;
  const int idx0 = blockIdx.x * 16;

  // phase 1: gather children h (f32) -> bf16 A-tile
  for (int task = tid; task < 16 * 128; task += 256) {
    int r = task >> 7, g = task & 127, col = g * 4;
    int gn = idx0 + r;
    int b = gn / nlevel, jl = gn % nlevel;
    int child = col >> 8;        // cols [0,256) = child0, [256,512) = child1
    int ccol  = col & 255;
    int te = e + 2 * jl + child;
    const float4 v = *(const float4*)(h_all + ((size_t)b * Tt + te) * Hh + ccol);
    U4b o;
    o.b[0] = (__bf16)v.x; o.b[1] = (__bf16)v.y;
    o.b[2] = (__bf16)v.z; o.b[3] = (__bf16)v.w;
    *(uint2*)(As + r * LV_AS + col) = o.q;
  }
  __syncthreads();

  // phase 2: WMMA GEMM 16x1280 (K=512). 8 waves x 10 N-tiles x 16 K-steps.
  {
    const int lane = tid & 31, wave = tid >> 5;
    const int nl  = lane & 15;
    const int hi8 = (lane >> 4) * 8;
    v8f acc[10];
#pragma unroll
    for (int j = 0; j < 10; ++j) acc[j] = v8f{0.f, 0.f, 0.f, 0.f, 0.f, 0.f, 0.f, 0.f};
    const __bf16* bp[10];
#pragma unroll
    for (int j = 0; j < 10; ++j) {
      int nt  = wave * 10 + j;
      int c16 = nt * 16 + nl;
      bp[j] = (nt < 32) ? (Uf + (size_t)c16 * 512)
                        : (Uiou + (size_t)(c16 - 512) * 512);
    }
    for (int ks = 0; ks < 16; ++ks) {
      int kb = ks * 32;
      ABu a;
      a.q[0] = *(const uint4*)(As + nl * LV_AS + kb + hi8);
      a.q[1] = *(const uint4*)(As + nl * LV_AS + kb + 16 + hi8);
#pragma unroll
      for (int j = 0; j < 10; ++j) {
        ABu bb;
        bb.q[0] = *(const uint4*)(bp[j] + kb + hi8);
        bb.q[1] = *(const uint4*)(bp[j] + kb + 16 + hi8);
        acc[j] = __builtin_amdgcn_wmma_f32_16x16x32_bf16(
            false, a.v, false, bb.v, (short)0, acc[j], false, false);
      }
    }
#pragma unroll
    for (int j = 0; j < 10; ++j) {
      int nt = wave * 10 + j;
#pragma unroll
      for (int r = 0; r < 8; ++r)
        Os[(r + hi8) * LV_OS + nt * 16 + nl] = acc[j][r];
    }
  }
  __syncthreads();

  // phase 3: f-gates + child-c reduction + iou gates
  for (int task = tid; task < 16 * 256; task += 256) {
    int r = task >> 8, j = task & 255;
    int gn = idx0 + r;
    int b = gn / nlevel, jl = gn % nlevel;
    int t   = s + jl;
    int te0 = e + 2 * jl, te1 = te0 + 1;
    size_t row = (size_t)b * Tt + t;
    size_t rc0 = ((size_t)b * Tt + te0) * Hh + j;
    size_t rc1 = ((size_t)b * Tt + te1) * Hh + j;
    float f0 = sigf(Os[r * LV_OS + j]       + Ufb[j]);
    float f1 = sigf(Os[r * LV_OS + 256 + j] + Ufb[256 + j]);
    float cred = f0 * c_all[rc0] + f1 * c_all[rc1];
    float i_ = Os[r * LV_OS + 512 + j]  + biou[j];
    float o_ = Os[r * LV_OS + 768 + j]  + biou[256 + j];
    float u_ = Os[r * LV_OS + 1024 + j] + biou[512 + j];
    float cn = sigf(i_) * tanhf(u_) + cred;
    float hh = sigf(o_) * tanhf(cn);
    c_all[row * Hh + j] = cn;
    h_all[row * Hh + j] = hh;
  }
}

// ------------------------------- root copy ----------------------------------
__global__ void root_kernel(const float* __restrict__ h_all,
                            const float* __restrict__ c_all,
                            float* __restrict__ out_tail) {
  int i = blockIdx.x * blockDim.x + threadIdx.x;
  const int BH = Bc * Hh;
  if (i >= 2 * BH) return;
  int part = i / BH, r = i % BH;
  int b = r / Hh, j = r % Hh;
  size_t src = (size_t)b * Tt * Hh + j;   // node 0 of tree b
  out_tail[i] = (part == 0) ? h_all[src] : c_all[src];
}

// ------------------------------- launcher -----------------------------------
extern "C" void kernel_launch(void* const* d_in, const int* in_sizes, int n_in,
                              void* d_out, int out_size, void* d_ws, size_t ws_size,
                              hipStream_t stream) {
  const int*   wordid = (const int*)d_in[0];
  const float* c0     = (const float*)d_in[3];
  const float* emb    = (const float*)d_in[6];
  const float* Wiou   = (const float*)d_in[7];   // [768][256]
  const float* Uiou   = (const float*)d_in[8];   // [768][512]
  const float* biou   = (const float*)d_in[9];   // [768]
  const float* Ufw    = (const float*)d_in[10];  // [512][512]
  const float* Ufb    = (const float*)d_in[11];  // [512]

  float* h_all = (float*)d_out;                  // tree_output (N,H)
  float* root  = h_all + (size_t)Nn * Hh;        // root_h (B,H) ++ root_c (B,H)

  char* ws = (char*)d_ws;
  float*  c_all   = (float*)ws;  ws += (size_t)Nn * Hh * sizeof(float);
  __bf16* emb_bf  = (__bf16*)ws; ws += (size_t)Vv * Ee * sizeof(__bf16);
  __bf16* Wiou_bf = (__bf16*)ws; ws += (size_t)768 * 256 * sizeof(__bf16);
  __bf16* Uf_bf   = (__bf16*)ws; ws += (size_t)512 * 512 * sizeof(__bf16);
  __bf16* Uiou_bf = (__bf16*)ws; ws += (size_t)768 * 512 * sizeof(__bf16);

  // weight / embedding conversion to bf16 (row-major: WMMA B-frags are
  // contiguous 16B runs along K)
  cvt_kernel<<<2048, 256, 0, stream>>>(emb,  emb_bf,  Vv * Ee);
  cvt_kernel<<<96,   256, 0, stream>>>(Wiou, Wiou_bf, 768 * 256);
  cvt_kernel<<<128,  256, 0, stream>>>(Ufw,  Uf_bf,   512 * 512);
  cvt_kernel<<<192,  256, 0, stream>>>(Uiou, Uiou_bf, 768 * 512);

  // leaves: 65536 rows / 16 per block
  size_t leaf_shmem = (size_t)16 * LF_AS * sizeof(__bf16) + (size_t)16 * LF_OS * sizeof(float);
  leaf_kernel<<<4096, 256, leaf_shmem, stream>>>(wordid, emb_bf, Wiou_bf, biou,
                                                 c0, h_all, c_all);

  // sequential levels, root-ward; stream order carries the dependency
  size_t lvl_shmem = (size_t)16 * LV_AS * sizeof(__bf16) + (size_t)16 * LV_OS * sizeof(float);
  for (int d = Dd - 2; d >= 0; --d) {
    int s = (1 << d) - 1, e = (1 << (d + 1)) - 1, n = 1 << d;
    int blocks = (Bc * n) / 16;
    level_kernel<<<blocks, 256, lvl_shmem, stream>>>(s, e, n, h_all, c_all,
                                                     Uf_bf, Uiou_bf, Ufb, biou);
  }

  root_kernel<<<(2 * Bc * Hh + 255) / 256, 256, 0, stream>>>(h_all, c_all, root);

  (void)in_sizes; (void)n_in; (void)out_size; (void)ws_size;
}
